// LSTM_44074954391592
// MI455X (gfx1250) — matrix operand order repaired and checked by reading, not compiled, not run on hardware
//
#include <hip/hip_runtime.h>
#include <hip/hip_bf16.h>

// LSTM: B=256, T=512, D=128, U=128. Persistent-workgroup design:
// 16 blocks x 1024 threads (32 waves); each block owns 16 batch rows for all
// 512 steps. Each wave owns one 16-column gate tile and keeps its 8 B-matrix
// WMMA fragments (bf16 weights) REGISTER-RESIDENT across the whole recurrence.
// Per step only the A fragments (x_t, h) are pulled from LDS (16 ds_load_b128)
// and 8 v_wmma_f32_16x16x32_bf16 run in two independent accumulator chains.

typedef __attribute__((ext_vector_type(16))) __bf16 v16bf;
typedef __attribute__((ext_vector_type(8)))  float  v8f;

constexpr int kB  = 256;   // batch
constexpr int kT  = 512;   // steps
constexpr int kD  = 128;   // input dim
constexpr int kU  = 128;   // units
constexpr int kNG = 4 * kU;      // 512 gate columns
constexpr int kBT = 16;          // batch rows per block (WMMA M)
constexpr int kTHREADS = 1024;   // 32 waves (wave32), one N-tile each

__device__ __forceinline__ unsigned short f2bf(float f) {
    unsigned int u = __float_as_uint(f);
    u += 0x7FFFu + ((u >> 16) & 1u);   // round-to-nearest-even
    return (unsigned short)(u >> 16);
}
__device__ __forceinline__ float sigmoidf(float x) {
    return 1.0f / (1.0f + __expf(-x));
}
__device__ __forceinline__ float tanh_fast(float x) {
    return 1.0f - 2.0f / (__expf(2.0f * x) + 1.0f);
}

union Frag { unsigned short s[16]; uint4 u4[2]; v16bf v; };

__global__ __launch_bounds__(kTHREADS)
void lstm_persistent(const float* __restrict__ x, const float* __restrict__ W,
                     const float* __restrict__ bias, float* __restrict__ out)
{
    __shared__ unsigned short xbuf[kBT * kD];   // x_t tile, bf16
    __shared__ unsigned short hbuf[kBT * kU];   // h tile, bf16
    __shared__ float          gates[kBT * kNG]; // f32 pre-activation gates

    const int tid   = threadIdx.x;
    const int lane  = tid & 31;
    const int wv    = tid >> 5;               // wave 0..31 == N-tile index
    const int bbase = blockIdx.x * kBT;

    // ---- this wave's gate-column and WMMA addressing constants ----
    const int ncol = lane & 15;
    const int n    = wv * 16 + ncol;          // global gate column 0..511
    const int boff = (lane >> 4) * 16;        // B: half-wave k offset
    const int am   = lane & 15;               // A row (M)
    const int aoff = (lane >> 4) * 16;        // A: half-wave k offset

    // ---- one-time: load this wave's 8 B fragments from global W, bf16 ----
    // B layout (16-bit, 32x16): lane L holds column n = L%16; element e holds
    // k = kt*32 + 16*(L/16) + e  (k contiguous over e).
    Frag fb[8];
    #pragma unroll
    for (int kt = 0; kt < 8; ++kt) {
        #pragma unroll
        for (int e = 0; e < 16; ++e) {
            fb[kt].s[e] = f2bf(W[(size_t)(kt * 32 + boff + e) * kNG + n]);
        }
    }
    const float bn = bias[n];

    // zero h buffer
    for (int idx = tid; idx < kBT * kU; idx += kTHREADS) hbuf[idx] = 0;

    // per-thread cell state: column eu, rows em + 8*j (2 per thread)
    const int eu = tid & (kU - 1);
    const int em = tid >> 7;                   // 0..7
    float c[2] = {0.f, 0.f};

    __syncthreads();

    for (int step = 0; step < kT; ++step) {
        // ---- stage x tile (16x128 f32 -> bf16 LDS), prefetch next step ----
        #pragma unroll
        for (int j = 0; j < 2; ++j) {
            int idx = tid + kTHREADS * j;               // 0..2047
            int m = idx >> 7, u = idx & 127;
            const float* src = x + ((size_t)(bbase + m) * kT + step) * kD + u;
            xbuf[idx] = f2bf(*src);
            if (step + 1 < kT) __builtin_prefetch(src + kD, 0, 1);
        }
        __syncthreads();

        // ---- GEMM: [x_t, h] (16x256) @ W tile (256x16), B in registers ----
        // Two independent accumulator chains (even/odd K-tiles) for XDL ILP.
        v8f acc0 = {};
        v8f acc1 = {};
        #pragma unroll
        for (int ktp = 0; ktp < 4; ++ktp) {
            const int kt0 = 2 * ktp, kt1 = 2 * ktp + 1;
            const unsigned short* asrc0 = (kt0 < 4) ? xbuf : hbuf;
            const unsigned short* asrc1 = (kt1 < 4) ? xbuf : hbuf;
            const int kk0 = (kt0 & 3) * 32 + aoff;
            const int kk1 = (kt1 & 3) * 32 + aoff;
            Frag a0, a1;
            a0.u4[0] = *(const uint4*)(asrc0 + am * kD + kk0);
            a0.u4[1] = *(const uint4*)(asrc0 + am * kD + kk0 + 8);
            a1.u4[0] = *(const uint4*)(asrc1 + am * kD + kk1);
            a1.u4[1] = *(const uint4*)(asrc1 + am * kD + kk1 + 8);
            acc0 = __builtin_amdgcn_wmma_f32_16x16x32_bf16(false, a0.v, false, fb[kt0].v,
                                                           (short)0, acc0, false, false);
            acc1 = __builtin_amdgcn_wmma_f32_16x16x32_bf16(false, a1.v, false, fb[kt1].v,
                                                           (short)0, acc1, false, false);
        }

        // ---- combine chains + bias, spill gates to LDS ----
        // C layout: M = r + 8*(lane/16), N = lane%16.
        #pragma unroll
        for (int r = 0; r < 8; ++r) {
            int m = r + ((lane >> 4) << 3);
            gates[m * kNG + n] = acc0[r] + acc1[r] + bn;
        }
        __syncthreads();

        // ---- elementwise LSTM cell update; h -> LDS (bf16) and global (f32) ----
        #pragma unroll
        for (int j = 0; j < 2; ++j) {
            int m = em + 8 * j;
            float gi = gates[m * kNG + eu];
            float gj = gates[m * kNG + kU + eu];
            float gf = gates[m * kNG + 2 * kU + eu];
            float go = gates[m * kNG + 3 * kU + eu];
            float nc = c[j] * sigmoidf(gf + 1.0f) + sigmoidf(gi) * tanh_fast(gj);
            c[j] = nc;
            float nh = tanh_fast(nc) * sigmoidf(go);
            hbuf[m * kU + eu] = f2bf(nh);
            out[((size_t)(bbase + m) * kT + step) * kU + eu] = nh;
        }
        __syncthreads();
    }
}

extern "C" void kernel_launch(void* const* d_in, const int* in_sizes, int n_in,
                              void* d_out, int out_size, void* d_ws, size_t ws_size,
                              hipStream_t stream) {
    const float* x    = (const float*)d_in[0];   // [256, 512, 128]
    const float* W    = (const float*)d_in[1];   // [256, 512]
    const float* bias = (const float*)d_in[2];   // [512]
    float* out = (float*)d_out;                  // [256, 512, 128]
    (void)in_sizes; (void)n_in; (void)out_size; (void)d_ws; (void)ws_size;

    dim3 grid(kB / kBT);     // 16 persistent workgroups
    dim3 block(kTHREADS);    // 1024 threads = 32 waves (wave32)
    lstm_persistent<<<grid, block, 0, stream>>>(x, W, bias, out);
}
